// Attn_35742717837754
// MI455X (gfx1250) — matrix operand order repaired and checked by reading, not compiled, not run on hardware
//
#include <hip/hip_runtime.h>

// ---------------------------------------------------------------------------
// Types for CDNA5 WMMA (wave32, v_wmma_f32_16x16x32_bf16)
// ---------------------------------------------------------------------------
typedef __attribute__((ext_vector_type(16))) __bf16 v16bf;
typedef __attribute__((ext_vector_type(8)))  __bf16 bf8;
typedef __attribute__((ext_vector_type(4)))  __bf16 bf4;
typedef __attribute__((ext_vector_type(8)))  float  v8f;

#define H_SIZE 1024
#define SEQ    2048
#define BATCH  32
#define SBLK   128             // s rows per workgroup (8 waves x 16)
#define STRE   (1024 + 8)     // enc LDS row stride (bf16 elems), 16B-multiple
#define STRW   (1024 + 8)     // W2-tile LDS row stride (bf16 elems)

#define ENC_BYTES   (SBLK * STRE * 2)          // 264192
#define WTILE_BYTES (16 * STRW * 2)            // 33024
#define PRE_OFF     (ENC_BYTES + WTILE_BYTES)  // f32 pre[1024]
#define V_OFF       (PRE_OFF + H_SIZE * 4)     // f32 v[1024]
#define SMEM_BYTES  (V_OFF + H_SIZE * 4)       // 305408

// ---------------------------------------------------------------------------
// Prep 1: W2 = W[:, H:2H] fp32 -> bf16, row-major [H][H]
// ---------------------------------------------------------------------------
__global__ __launch_bounds__(256) void prep_w2_bf16(const float* __restrict__ W,
                                                    __bf16* __restrict__ Wbf) {
    int idx = blockIdx.x * 256 + threadIdx.x;      // 0 .. 1048575
    int h = idx >> 10;
    int k = idx & 1023;
    Wbf[idx] = (__bf16)W[(size_t)h * (2 * H_SIZE) + H_SIZE + k];
}

// ---------------------------------------------------------------------------
// Prep 2: pre[b][h] = hidden[b] . W1[h] + bias[h]   (W1 = W[:, 0:H])
// ---------------------------------------------------------------------------
__global__ __launch_bounds__(256) void prep_pre(const float* __restrict__ hidden,
                                                const float* __restrict__ W,
                                                const float* __restrict__ bias,
                                                float* __restrict__ pre) {
    int idx = blockIdx.x * 256 + threadIdx.x;      // 0 .. 32767
    int b = idx >> 10;
    int h = idx & 1023;
    const float4* hv = (const float4*)(hidden + (size_t)b * H_SIZE);
    const float4* wv = (const float4*)(W + (size_t)h * (2 * H_SIZE));
    float acc = 0.f;
#pragma unroll 4
    for (int i = 0; i < H_SIZE / 4; ++i) {
        float4 x = hv[i];
        float4 y = wv[i];
        acc += x.x * y.x + x.y * y.y + x.z * y.z + x.w * y.w;
    }
    pre[idx] = acc + bias[h];
}

// ---------------------------------------------------------------------------
// Main: fused  tanh(enc @ W2^T + pre) . v  -> scores[b][s]
// grid = (SEQ/SBLK, BATCH), block = 256 (8 waves); wave w owns s-tile w.
// All waves share each 16-row W2 h-tile, staged global->LDS with
// global_load_async_to_lds_b128 (ASYNCcnt) once per workgroup.
// D = A x B with A = W2 tile (16h x 32k), B = enc^T tile (32k x 16s).
// ---------------------------------------------------------------------------
__global__ __launch_bounds__(256) void attn_scores(const float*  __restrict__ enc,
                                                   const __bf16* __restrict__ Wbf,
                                                   const float*  __restrict__ pre,
                                                   const float*  __restrict__ v,
                                                   float*        __restrict__ scores) {
    extern __shared__ char smem[];
    __bf16* encS  = (__bf16*)smem;                       // SBLK x STRE bf16
    __bf16* wtile = (__bf16*)(smem + ENC_BYTES);         // 16 x STRW bf16
    float*  pre_s = (float*)(smem + PRE_OFF);
    float*  v_s   = (float*)(smem + V_OFF);

    const int tid    = threadIdx.x;
    const int b      = blockIdx.y;
    const int sbase0 = blockIdx.x * SBLK;

    // ---- stage enc strip (fp32 HBM -> bf16 LDS), read HBM exactly once ----
    for (int i = tid; i < SBLK * (H_SIZE / 4); i += 256) {
        int row = i >> 8;                 // 0..127
        int c4  = i & 255;                // float4 index within row
        size_t g = ((size_t)(sbase0 + row) * BATCH + b) * H_SIZE + c4 * 4;
        float4 x = *(const float4*)(enc + g);
        bf4 y = { (__bf16)x.x, (__bf16)x.y, (__bf16)x.z, (__bf16)x.w };
        *(bf4*)(encS + (size_t)row * STRE + c4 * 4) = y;
    }
    for (int i = tid; i < H_SIZE; i += 256) {
        pre_s[i] = pre[(size_t)b * H_SIZE + i];
        v_s[i]   = v[i];
    }

    const int lane = tid & 31;
    const int wid  = tid >> 5;           // s-tile index 0..7
    const int hi   = lane >> 4;          // lane half (K-group select)
    const int lr   = lane & 15;          // row index within fragment

    const __bf16* encRow = encS + (size_t)(wid * 16 + lr) * STRE;
    const __bf16* wrow   = wtile + (size_t)lr * STRW;
    const unsigned long long wbase = (unsigned long long)(uintptr_t)Wbf;

    float score = 0.f;
    __syncthreads();

    for (int ht = 0; ht < H_SIZE / 16; ++ht) {
        // ---- stage 16x1024 bf16 W2 tile: contiguous 32 KiB, async to LDS ----
        // chunk i (b128): global off = ht*32768 + i*16 ; LDS = WTILE + row*pitch
#pragma unroll
        for (int r = 0; r < 8; ++r) {
            unsigned i    = (unsigned)tid + r * 256u;          // 0..2047
            unsigned row  = i >> 7;                            // 0..15
            unsigned cc   = i & 127u;                          // b128 within row
            unsigned ldsb = (unsigned)ENC_BYTES + row * (STRW * 2u) + cc * 16u;
            unsigned goff = (unsigned)ht * 32768u + i * 16u;
            asm volatile("global_load_async_to_lds_b128 %0, %1, %2"
                         :: "v"(ldsb), "v"(goff), "s"(wbase)
                         : "memory");
        }
        asm volatile("s_wait_asynccnt 0x0" ::: "memory");
        __syncthreads();

        const int hbase = ht * 16;
        v8f c = {0.f, 0.f, 0.f, 0.f, 0.f, 0.f, 0.f, 0.f};

#pragma unroll 4
        for (int k = 0; k < H_SIZE; k += 32) {
            // A frag: lanes 0-15 hold K {0..7,16..23}; lanes 16-31 {8..15,24..31}
            bf8 a0 = *(const bf8*)(wrow + k + hi * 8);
            bf8 a1 = *(const bf8*)(wrow + k + 16 + hi * 8);
            v16bf a = __builtin_shufflevector(a0, a1,
                0,1,2,3,4,5,6,7,8,9,10,11,12,13,14,15);
            // B frag: lanes 0-15 hold K {0..7,16..23}; lanes 16-31 {8..15,24..31}
            bf8 b0 = *(const bf8*)(encRow + k + hi * 8);
            bf8 b1 = *(const bf8*)(encRow + k + 16 + hi * 8);
            v16bf bb = __builtin_shufflevector(b0, b1,
                0,1,2,3,4,5,6,7,8,9,10,11,12,13,14,15);

            c = __builtin_amdgcn_wmma_f32_16x16x32_bf16(
                    false, a, false, bb, (short)0, c, false, false);
        }

        // fused epilogue: score[s] += tanh(d + pre[h]) * v[h]
        // C layout: VGPR j -> row (j + 8*hi), col = lane%16
#pragma unroll
        for (int j = 0; j < 8; ++j) {
            int h = hbase + j + 8 * hi;
            float x = c[j] + pre_s[h];
            x = fminf(fmaxf(x, -15.f), 15.f);           // branch-free tanh
            float ex = __expf(2.f * x);
            float t  = (ex - 1.f) * __builtin_amdgcn_rcpf(ex + 1.f);
            score += t * v_s[h];
        }

        __syncthreads();   // all waves done with wtile before restaging
    }

    // combine lane L with lane L+16 (same s column, complementary rows)
    score += __shfl_xor(score, 16, 32);
    if (lane < 16) {
        scores[(size_t)b * SEQ + sbase0 + wid * 16 + lane] = score;
    }
}

// ---------------------------------------------------------------------------
// Softmax over S per batch row.
// ---------------------------------------------------------------------------
__global__ __launch_bounds__(256) void softmax_rows(const float* __restrict__ scores,
                                                    float* __restrict__ out) {
    __shared__ float sm[SEQ];
    __shared__ float red[256];
    const int b = blockIdx.x;
    const int tid = threadIdx.x;
    const float* p = scores + (size_t)b * SEQ;

    float lmax = -3.402823e38f;
    for (int i = tid; i < SEQ; i += 256) {
        float s = p[i];
        sm[i] = s;
        lmax = fmaxf(lmax, s);
    }
    red[tid] = lmax;
    __syncthreads();
    for (int off = 128; off > 0; off >>= 1) {
        if (tid < off) red[tid] = fmaxf(red[tid], red[tid + off]);
        __syncthreads();
    }
    const float m = red[0];
    __syncthreads();

    float lsum = 0.f;
    for (int i = tid; i < SEQ; i += 256) {
        float e = __expf(sm[i] - m);
        sm[i] = e;
        lsum += e;
    }
    red[tid] = lsum;
    __syncthreads();
    for (int off = 128; off > 0; off >>= 1) {
        if (tid < off) red[tid] += red[tid + off];
        __syncthreads();
    }
    const float inv = 1.f / red[0];
    for (int i = tid; i < SEQ; i += 256)
        out[(size_t)b * SEQ + i] = sm[i] * inv;
}

// ---------------------------------------------------------------------------
// Host-side launcher
// ---------------------------------------------------------------------------
extern "C" void kernel_launch(void* const* d_in, const int* in_sizes, int n_in,
                              void* d_out, int out_size, void* d_ws, size_t ws_size,
                              hipStream_t stream) {
    const float* hidden = (const float*)d_in[0];   // [32, 1024]
    const float* enc    = (const float*)d_in[1];   // [2048, 32, 1024]
    const float* W      = (const float*)d_in[2];   // [1024, 2048]
    const float* bias   = (const float*)d_in[3];   // [1024]
    const float* v      = (const float*)d_in[4];   // [1024]
    float* out          = (float*)d_out;           // [32, 1, 2048]

    // Workspace: Wbf bf16[1024*1024] (2 MiB) | pre f32[32*1024] | scores f32[32*2048]
    char* ws = (char*)d_ws;
    __bf16* Wbf    = (__bf16*)ws;
    float*  pre    = (float*)(ws + (2u << 20));
    float*  scores = (float*)(ws + (2u << 20) + (128u << 10));

    prep_w2_bf16<<<(H_SIZE * H_SIZE) / 256, 256, 0, stream>>>(W, Wbf);
    prep_pre<<<(BATCH * H_SIZE) / 256, 256, 0, stream>>>(hidden, W, bias, pre);

    attn_scores<<<dim3(SEQ / SBLK, BATCH), 256, SMEM_BYTES, stream>>>(
        enc, Wbf, pre, v, scores);

    softmax_rows<<<BATCH, 256, 0, stream>>>(scores, out);
}
